// TemporalGNNModel_515396076301
// MI455X (gfx1250) — matrix operand order repaired and checked by reading, not compiled, not run on hardware
//
#include <hip/hip_runtime.h>
#include <hip/hip_fp16.h>

// ---------------------------------------------------------------------------
// TemporalGNN (GCN + LSTM + FC) for MI455X / gfx1250, wave32, f16 WMMA path.
// ---------------------------------------------------------------------------

typedef __attribute__((ext_vector_type(16))) _Float16 v16h;
typedef __attribute__((ext_vector_type(8)))  _Float16 v8h;
typedef __attribute__((ext_vector_type(4)))  _Float16 v4h;
typedef __attribute__((ext_vector_type(8)))  float    v8f;
typedef __attribute__((ext_vector_type(4)))  float    v4f;

#define T_STEPS 50
#define N_NODES 20000
#define F_IN    64
#define H_DIM   64
#define G4H     256        // 4*H
#define E_EDGES 640000
#define LSTRIDE 72         // f16 LDS row stride: 144B = 16B-aligned, conflict-free

static __device__ __forceinline__ void fadd_atomic(float* p, float v) {
  __hip_atomic_fetch_add(p, v, __ATOMIC_RELAXED, __HIP_MEMORY_SCOPE_AGENT);
}
static __device__ __forceinline__ float sigm(float x) {
  return 1.0f / (1.0f + __expf(-x));
}
static __device__ __forceinline__ float tanh_(float x) {
  x = fminf(fmaxf(x, -15.0f), 15.0f);
  float e = __expf(2.0f * x);
  return (e - 1.0f) / (e + 1.0f);
}
static __device__ __forceinline__ v16h cat16(v8h lo, v8h hi) {
  v16h r;
#pragma unroll
  for (int i = 0; i < 8; ++i) { r[i] = lo[i]; r[i + 8] = hi[i]; }
  return r;
}

// A fragment (16x32 f16) from LDS tile [16][LSTRIDE]:
//   lane l<16 : row=l,    K = {kb+0..7,  kb+16..23}
//   lane l>=16: row=l-16, K = {kb+8..15, kb+24..31}
static __device__ __forceinline__ v16h load_a_frag(const _Float16* tile, int kb) {
  const int l = threadIdx.x & 31;
  const _Float16* p = tile + (l & 15) * LSTRIDE + kb + ((l & 16) ? 8 : 0);
  return cat16(*(const v8h*)p, *(const v8h*)(p + 16));
}

// B fragment (32x16 f16) from row-major [n][k] LDS tile:
//   lane l: col n = (l&15) + 16*nt,  K = kb + (l<16 ? {0..7,16..23} : {8..15,24..31})
static __device__ __forceinline__ v16h load_b_frag(const _Float16* wT, int nt, int kb) {
  const int l = threadIdx.x & 31;
  const _Float16* p = wT + ((l & 15) + nt * 16) * LSTRIDE + kb + ((l & 16) ? 8 : 0);
  return cat16(*(const v8h*)p, *(const v8h*)(p + 16));
}

// ---------------------------------------------------------------------------
// Setup kernels
// ---------------------------------------------------------------------------
__global__ void init_state_kernel(float* deg, float* c, _Float16* h16) {
  int idx = blockIdx.x * blockDim.x + threadIdx.x;            // over N*H
  if (idx < N_NODES * H_DIM) { c[idx] = 0.0f; h16[idx] = (_Float16)0.0f; }
  if (idx < N_NODES) deg[idx] = 1.0f;                         // self-loop
}

__global__ void zero_agg_kernel(float* agg) {
  size_t idx = (size_t)blockIdx.x * blockDim.x + threadIdx.x; // over T*N*H/4
  ((v4f*)agg)[idx] = (v4f){0.f, 0.f, 0.f, 0.f};
}

__global__ void deg_accum_kernel(const long long* __restrict__ ei, float* deg) {
  int e = blockIdx.x * blockDim.x + threadIdx.x;
  if (e < E_EDGES) fadd_atomic(&deg[(int)ei[E_EDGES + e]], 1.0f);
}

__global__ void dinv_kernel(const float* __restrict__ deg, float* dinv) {
  int n = blockIdx.x * blockDim.x + threadIdx.x;
  if (n < N_NODES) { float d = deg[n]; dinv[n] = d > 0.f ? __frsqrt_rn(d) : 0.f; }
}

__global__ void edge_norm_kernel(const long long* __restrict__ ei,
                                 const float* __restrict__ dinv, float* norm) {
  int e = blockIdx.x * blockDim.x + threadIdx.x;
  if (e < E_EDGES) norm[e] = dinv[(int)ei[e]] * dinv[(int)ei[E_EDGES + e]];
}

// ---------------------------------------------------------------------------
// GCN transform: xw[t*N+n, :] = f16( x[t,n,:] @ gcn_w )   (WMMA f16->f32)
// one wave = 16 rows x 64 cols; 8 waves/block
// ---------------------------------------------------------------------------
__global__ void __launch_bounds__(256)
gcn_transform_kernel(const float* __restrict__ x, const float* __restrict__ gcn_w,
                     _Float16* __restrict__ xw) {
  __shared__ __align__(16) _Float16 s_wT[H_DIM * LSTRIDE];   // gcn_w transposed [n][k]
  __shared__ __align__(16) _Float16 s_x[8][16 * LSTRIDE];
  const int tid = threadIdx.x, wave = tid >> 5, lane = tid & 31;

  for (int i = tid; i < H_DIM * H_DIM; i += 256) {            // s_wT[n][k] = gw[k][n]
    int k = i >> 6, n = i & 63;
    s_wT[n * LSTRIDE + k] = (_Float16)gcn_w[i];
  }
  const long rowBase = ((long)blockIdx.x * 8 + wave) * 16;
  const long totalRows = (long)T_STEPS * N_NODES;
  const bool active = rowBase < totalRows;
  if (active) {
    for (int i = lane; i < 16 * 16; i += 32) {                // 16 rows x 4-float chunks
      int r = i >> 4, cb = (i & 15) * 4;
      v4f v = *(const v4f*)&x[(size_t)(rowBase + r) * F_IN + cb];
      v4h h4;
#pragma unroll
      for (int j = 0; j < 4; ++j) h4[j] = (_Float16)v[j];
      *(v4h*)&s_x[wave][r * LSTRIDE + cb] = h4;
    }
  }
  __syncthreads();
  if (!active) return;

  v8f acc[4] = {};
#pragma unroll
  for (int kb = 0; kb < F_IN; kb += 32) {
    v16h a = load_a_frag(s_x[wave], kb);
#pragma unroll
    for (int nt = 0; nt < 4; ++nt) {
      v16h b = load_b_frag(s_wT, nt, kb);
      acc[nt] = __builtin_amdgcn_wmma_f32_16x16x32_f16(
          false, a, false, b, (short)0, acc[nt], false, false);
    }
  }
  const int ln = lane & 15, rofs = (lane & 16) ? 8 : 0;
#pragma unroll
  for (int nt = 0; nt < 4; ++nt)
#pragma unroll
    for (int j = 0; j < 8; ++j)
      xw[(size_t)(rowBase + j + rofs) * H_DIM + nt * 16 + ln] = (_Float16)acc[nt][j];
}

// ---------------------------------------------------------------------------
// Edge aggregation: agg[t,dst,:] += norm * xw[t,src,:]   (one wave per edge,
// 2 cols/lane, hardware f32 atomics; coalesced 128B rows)
// ---------------------------------------------------------------------------
__global__ void __launch_bounds__(256)
aggregate_kernel(const long long* __restrict__ ei, const float* __restrict__ norm,
                 const _Float16* __restrict__ xw, float* __restrict__ agg) {
  const int t = blockIdx.y;
  const long flat = (long)blockIdx.x * 256 + threadIdx.x;
  const long e = flat >> 5;
  const int ln = (int)(flat & 31);
  if (e >= E_EDGES) return;
  __builtin_prefetch(ei + e + 2048, 0, 1);                    // global_prefetch_b8
  const int s = (int)ei[e], d = (int)ei[E_EDGES + e];
  const float w = norm[e];
  const _Float16* ps = xw + ((size_t)t * N_NODES + s) * H_DIM + ln * 2;
  float* pd = agg + ((size_t)t * N_NODES + d) * H_DIM + ln * 2;
  fadd_atomic(pd,     (float)ps[0] * w);
  fadd_atomic(pd + 1, (float)ps[1] * w);
}

// seq = relu(agg + dinv^2 * xw + b), written back into xw (elementwise, in place)
__global__ void seq_finalize_kernel(const float* __restrict__ agg,
                                    const float* __restrict__ dinv,
                                    const float* __restrict__ gcn_b,
                                    _Float16* __restrict__ xw) {
  size_t idx = (size_t)blockIdx.x * blockDim.x + threadIdx.x; // over T*N*H
  int h = (int)(idx & 63);
  int n = (int)((idx >> 6) % N_NODES);
  float di = dinv[n];
  float v = agg[idx] + di * di * (float)xw[idx] + gcn_b[h];
  xw[idx] = (_Float16)fmaxf(v, 0.0f);
}

// ---------------------------------------------------------------------------
// Fused LSTM step: gates = seq_t@Wih^T + h@Whh^T + b, cell update in registers.
// One wave = 16 rows x 256 gate cols (16 accumulator tiles); 4 waves/block.
// Gate cols for a given (row,h) live in the SAME lane/VGPR across tiles
// {ht, 4+ht, 8+ht, 12+ht} -> fully lane-local cell math. In-place h/c update
// is safe (row-local dependence).
// ---------------------------------------------------------------------------
__global__ void __launch_bounds__(128)
lstm_step_kernel(const _Float16* __restrict__ seq_t, _Float16* __restrict__ h16,
                 float* __restrict__ c, const float* __restrict__ w_ih,
                 const float* __restrict__ w_hh, const float* __restrict__ b_ih,
                 const float* __restrict__ b_hh) {
  __shared__ __align__(16) _Float16 s_w[G4H * LSTRIDE];       // 36.9 KB (reused)
  __shared__ __align__(16) _Float16 s_a[4][16 * LSTRIDE];
  __shared__ __align__(16) _Float16 s_h[4][16 * LSTRIDE];
  __shared__ float s_bias[G4H];

  const int tid = threadIdx.x, wave = tid >> 5, lane = tid & 31;
  const int rowBase = (blockIdx.x * 4 + wave) * 16;
  const bool active = rowBase < N_NODES;

  for (int i = tid; i < G4H; i += 128) s_bias[i] = b_ih[i] + b_hh[i];
  if (active) {
    for (int i = lane; i < 16 * 8; i += 32) {                 // 16B chunks
      int r = i >> 3, cb = (i & 7) * 8;
      size_t g = (size_t)(rowBase + r) * H_DIM + cb;
      *(v8h*)&s_a[wave][r * LSTRIDE + cb] = *(const v8h*)&seq_t[g];
      *(v8h*)&s_h[wave][r * LSTRIDE + cb] = *(const v8h*)&h16[g];
    }
  }
  for (int i = tid; i < G4H * 16; i += 128) {                 // stage w_ih as f16
    int n = i >> 4, cb = (i & 15) * 4;
    v4f w = *(const v4f*)&w_ih[n * H_DIM + cb];
    v4h h4;
#pragma unroll
    for (int j = 0; j < 4; ++j) h4[j] = (_Float16)w[j];
    *(v4h*)&s_w[n * LSTRIDE + cb] = h4;
  }
  __syncthreads();

  v8f acc[16] = {};
  if (active) {
#pragma unroll
    for (int kb = 0; kb < H_DIM; kb += 32) {
      v16h a = load_a_frag(s_a[wave], kb);
#pragma unroll
      for (int nt = 0; nt < 16; ++nt) {
        v16h b = load_b_frag(s_w, nt, kb);
        acc[nt] = __builtin_amdgcn_wmma_f32_16x16x32_f16(
            false, a, false, b, (short)0, acc[nt], false, false);
      }
    }
  }
  __syncthreads();
  for (int i = tid; i < G4H * 16; i += 128) {                 // restage with w_hh
    int n = i >> 4, cb = (i & 15) * 4;
    v4f w = *(const v4f*)&w_hh[n * H_DIM + cb];
    v4h h4;
#pragma unroll
    for (int j = 0; j < 4; ++j) h4[j] = (_Float16)w[j];
    *(v4h*)&s_w[n * LSTRIDE + cb] = h4;
  }
  __syncthreads();
  if (!active) return;

#pragma unroll
  for (int kb = 0; kb < H_DIM; kb += 32) {
    v16h a = load_a_frag(s_h[wave], kb);
#pragma unroll
    for (int nt = 0; nt < 16; ++nt) {
      v16h b = load_b_frag(s_w, nt, kb);
      acc[nt] = __builtin_amdgcn_wmma_f32_16x16x32_f16(
          false, a, false, b, (short)0, acc[nt], false, false);
    }
  }

  const int ln = lane & 15, rofs = (lane & 16) ? 8 : 0;
#pragma unroll
  for (int ht = 0; ht < 4; ++ht) {
#pragma unroll
    for (int j = 0; j < 8; ++j) {
      int row = rowBase + j + rofs;
      int h = ht * 16 + ln;
      float gi = acc[ht][j]      + s_bias[h];
      float gf = acc[4 + ht][j]  + s_bias[64 + h];
      float gg = acc[8 + ht][j]  + s_bias[128 + h];
      float go = acc[12 + ht][j] + s_bias[192 + h];
      size_t p = (size_t)row * H_DIM + h;
      float cn = sigm(gf) * c[p] + sigm(gi) * tanh_(gg);
      c[p] = cn;
      h16[p] = (_Float16)(sigm(go) * tanh_(cn));
    }
  }
}

__global__ void fc_kernel(const _Float16* __restrict__ h16,
                          const float* __restrict__ fc_w,
                          const float* __restrict__ fc_b, float* __restrict__ out) {
  int n = blockIdx.x * blockDim.x + threadIdx.x;
  if (n >= N_NODES) return;
  float s = fc_b[0];
#pragma unroll
  for (int k = 0; k < H_DIM; ++k) s += (float)h16[(size_t)n * H_DIM + k] * fc_w[k];
  out[n] = s;
}

// ---------------------------------------------------------------------------
extern "C" void kernel_launch(void* const* d_in, const int* in_sizes, int n_in,
                              void* d_out, int out_size, void* d_ws, size_t ws_size,
                              hipStream_t stream) {
  (void)in_sizes; (void)n_in; (void)out_size; (void)ws_size;
  const float*     x   = (const float*)d_in[0];
  const long long* ei  = (const long long*)d_in[1];
  const float*     gw  = (const float*)d_in[2];
  const float*     gb  = (const float*)d_in[3];
  const float*     wih = (const float*)d_in[4];
  const float*     whh = (const float*)d_in[5];
  const float*     bih = (const float*)d_in[6];
  const float*     bhh = (const float*)d_in[7];
  const float*     fcw = (const float*)d_in[8];
  const float*     fcb = (const float*)d_in[9];
  float* out = (float*)d_out;

  auto au = [](size_t v) { return (v + 255) & ~(size_t)255; };
  char* p = (char*)d_ws;
  float*    deg  = (float*)p;    p += au((size_t)N_NODES * 4);
  float*    dinv = (float*)p;    p += au((size_t)N_NODES * 4);
  float*    norm = (float*)p;    p += au((size_t)E_EDGES * 4);
  _Float16* xw   = (_Float16*)p; p += au((size_t)T_STEPS * N_NODES * H_DIM * 2);
  float*    agg  = (float*)p;    p += au((size_t)T_STEPS * N_NODES * H_DIM * 4);
  _Float16* h16  = (_Float16*)p; p += au((size_t)N_NODES * H_DIM * 2);
  float*    c    = (float*)p;    p += au((size_t)N_NODES * H_DIM * 4);

  const size_t TNH = (size_t)T_STEPS * N_NODES * H_DIM;       // 64,000,000

  init_state_kernel<<<(N_NODES * H_DIM + 255) / 256, 256, 0, stream>>>(deg, c, h16);
  zero_agg_kernel<<<(unsigned)(TNH / 4 / 256), 256, 0, stream>>>(agg);
  deg_accum_kernel<<<(E_EDGES + 255) / 256, 256, 0, stream>>>(ei, deg);
  dinv_kernel<<<(N_NODES + 255) / 256, 256, 0, stream>>>(deg, dinv);
  edge_norm_kernel<<<(E_EDGES + 255) / 256, 256, 0, stream>>>(ei, dinv, norm);

  const long totalRows = (long)T_STEPS * N_NODES;
  gcn_transform_kernel<<<(unsigned)((totalRows / 16 + 7) / 8), 256, 0, stream>>>(x, gw, xw);

  dim3 agrid((E_EDGES * 32 + 255) / 256, T_STEPS);
  aggregate_kernel<<<agrid, 256, 0, stream>>>(ei, norm, xw, agg);
  seq_finalize_kernel<<<(unsigned)(TNH / 256), 256, 0, stream>>>(agg, dinv, gb, xw);

  const unsigned lstmBlocks = (N_NODES + 63) / 64;            // 4 waves x 16 rows
  for (int t = 0; t < T_STEPS; ++t) {
    lstm_step_kernel<<<lstmBlocks, 128, 0, stream>>>(
        xw + (size_t)t * N_NODES * H_DIM, h16, c, wih, whh, bih, bhh);
  }
  fc_kernel<<<(N_NODES + 255) / 256, 256, 0, stream>>>(h16, fcw, fcb, out);
}
